// global_net_19413252178584
// MI455X (gfx1250) — compile-verified, matrix operand order checked by reading, not run on hardware
//
#include <hip/hip_runtime.h>
#include <hip/hip_bf16.h>

// ---------------- problem constants ----------------
#define BB   2
#define NPTS 1024
#define CD   64
#define SS   (NPTS * NPTS)        // 1048576
#define KTOP (SS / 6)             // 174762 kept entries per batch

typedef __attribute__((ext_vector_type(16))) __bf16 v16bf;
typedef __attribute__((ext_vector_type(8)))  __bf16 v8bf;
typedef __attribute__((ext_vector_type(8)))  float  v8f;
typedef __attribute__((__vector_size__(4 * sizeof(int)))) int gvec4i;

// ---- CDNA5 async global->LDS path (guarded; falls back to sync copies) ----
#if defined(__gfx1250__) && \
    __has_builtin(__builtin_amdgcn_global_load_async_to_lds_b128) && \
    __has_builtin(__builtin_amdgcn_s_wait_asynccnt)
#define USE_ASYNC 1
#else
#define USE_ASYNC 0
#endif

#define AS1 __attribute__((address_space(1)))
#define AS3 __attribute__((address_space(3)))

// copy 32 bytes (16 bf16) global -> LDS
__device__ __forceinline__ void cp32_g2l(const __bf16* g, __bf16* l) {
#if USE_ASYNC
  __builtin_amdgcn_global_load_async_to_lds_b128((AS1 gvec4i*)g,       (AS3 gvec4i*)l,       0, 0);
  __builtin_amdgcn_global_load_async_to_lds_b128((AS1 gvec4i*)(g + 8), (AS3 gvec4i*)(l + 8), 0, 0);
#else
  *(uint4*)l       = *(const uint4*)g;
  *(uint4*)(l + 8) = *(const uint4*)(g + 8);
#endif
}

template <int N> __device__ __forceinline__ void wait_async() {
#if USE_ASYNC
  __builtin_amdgcn_s_wait_asynccnt(N);
#endif
}

// ======================================================================
// K1: per-row mean + bf16 copy of x
// ======================================================================
__global__ void k_rowstats(const float* __restrict__ x, __bf16* __restrict__ xbf,
                           float* __restrict__ mean) {
  int row = blockIdx.x * blockDim.x + threadIdx.x;        // 0 .. B*N-1
  if (row >= BB * NPTS) return;
  const float* xr = x + (long)row * CD;
  float s = 0.f;
  #pragma unroll 8
  for (int c = 0; c < CD; ++c) {
    float v = xr[c];
    s += v;
    xbf[(long)row * CD + c] = (__bf16)v;
  }
  mean[row] = s * (1.0f / CD);
}

// ======================================================================
// NT WMMA GEMM: C[m,n] = sum_k A[m,k]*B[n,k]   (A,B bf16 row-major [.,K],
// C f32 row-major [M,N]).  Block tile 128x128, 8 waves, K step 32,
// double-buffered LDS fed by async global->LDS b128 copies.
// ======================================================================
#define LDSTR 40   // bf16 LDS row stride (32 data + 8 pad) -> 80B, 16B aligned

__device__ __forceinline__ v16bf load_frag(const __bf16* base, int lane) {
  // 16-bit A/B fragment (ISA 7.12.2): lane l<16 -> row l, K {0..7,16..23};
  // lane l>=16 -> row l-16, K {8..15,24..31}
  int r  = lane & 15;
  int kh = (lane >> 4) << 3;          // 0 or 8
  const __bf16* p = base + r * LDSTR + kh;
  v8bf lo = *(const v8bf*)(p);
  v8bf hi = *(const v8bf*)(p + 16);
  return __builtin_shufflevector(lo, hi, 0,1,2,3,4,5,6,7,8,9,10,11,12,13,14,15);
}

__global__ __launch_bounds__(256)
void k_gemm_nt(const __bf16* __restrict__ A, const __bf16* __restrict__ Bm,
               float* __restrict__ Cm, int M, int Nc, int K,
               long sA, long sB, long sC) {
  __shared__ __bf16 As[2][128 * LDSTR];
  __shared__ __bf16 Bs[2][128 * LDSTR];
  int b = blockIdx.z;
  const __bf16* Ab = A  + (long)b * sA;
  const __bf16* Bb = Bm + (long)b * sB;
  float*        Cb = Cm + (long)b * sC;

  int tn = blockIdx.x * 128;
  int tm = blockIdx.y * 128;
  int t    = threadIdx.x;
  int lane = t & 31;
  int w    = t >> 5;                  // wave 0..7
  int wm   = (w & 3) * 32;            // wave row offset in block tile
  int wn   = (w >> 2) * 64;           // wave col offset in block tile

  // staging map: thread -> (row, 32B column segment) of the 128x32 tiles
  int r  = t >> 1;                    // 0..127
  int cc = (t & 1) * 16;              // bf16 col 0 or 16

  v8f acc[2][4];
  #pragma unroll
  for (int i = 0; i < 2; ++i)
    #pragma unroll
    for (int j = 0; j < 4; ++j)
      #pragma unroll
      for (int e = 0; e < 8; ++e) acc[i][j][e] = 0.f;

  auto load_tile = [&](int buf, int k0) {
    cp32_g2l(Ab + (long)(tm + r) * K + k0 + cc, &As[buf][r * LDSTR + cc]);
    cp32_g2l(Bb + (long)(tn + r) * K + k0 + cc, &Bs[buf][r * LDSTR + cc]);
  };

  int nk = K >> 5;
  load_tile(0, 0);                    // prologue: 4 async b128 per thread

  for (int i = 0; i < nk; ++i) {
    int cur = i & 1;
    if (i + 1 < nk) {
      load_tile(cur ^ 1, (i + 1) << 5);
      wait_async<4>();                // previous tile's copies have landed
    } else {
      wait_async<0>();
    }
    __syncthreads();

    v16bf af[2], bfr[4];
    #pragma unroll
    for (int ii = 0; ii < 2; ++ii) af[ii]  = load_frag(&As[cur][(wm + 16 * ii) * LDSTR], lane);
    #pragma unroll
    for (int jj = 0; jj < 4; ++jj) bfr[jj] = load_frag(&Bs[cur][(wn + 16 * jj) * LDSTR], lane);

    #pragma unroll
    for (int ii = 0; ii < 2; ++ii)
      #pragma unroll
      for (int jj = 0; jj < 4; ++jj)
        acc[ii][jj] = __builtin_amdgcn_wmma_f32_16x16x32_bf16(
            false, af[ii], false, bfr[jj], (short)0, acc[ii][jj], false, false);
    __syncthreads();                  // buf[cur] free for overwrite next iter
  }

  // C/D layout: lane = {N=lane&15, Mhalf=lane>>4}, VGPR v -> M = v + 8*Mhalf
  // int offsets only (per-batch C is 4MB -> fits easily in 32 bits)
  int nlo = lane & 15, hiM = lane >> 4;
  float* Cw = Cb + (tm + wm + 8 * hiM) * Nc + tn + wn + nlo;
  #pragma unroll
  for (int i = 0; i < 2; ++i)
    #pragma unroll
    for (int j = 0; j < 4; ++j) {
      int base = (16 * i) * Nc + 16 * j;
      #pragma unroll
      for (int v = 0; v < 8; ++v)
        Cw[base + v * Nc] = acc[i][j][v];
    }
}

// ======================================================================
// K3a: diag(G) and 1/std per row
// ======================================================================
__global__ void k_rowaux(const float* __restrict__ G, const float* __restrict__ mean,
                         float* __restrict__ diag, float* __restrict__ rstd) {
  int row = blockIdx.x * blockDim.x + threadIdx.x;
  if (row >= BB * NPTS) return;
  int b = row >> 10, i = row & (NPTS - 1);
  float d = G[(long)b * SS + (long)i * NPTS + i];
  float m = mean[row];
  float var = d - (float)CD * m * m;
  diag[row] = d;
  rstd[row] = rsqrtf(fmaxf(var, 0.f) + 1e-20f);
}

// ======================================================================
// K3b: E (euclidean) and Corr from G, emitted as bf16 GEMM operands
// ======================================================================
__global__ void k_ecorr(const float* __restrict__ G, const float* __restrict__ diag,
                        const float* __restrict__ mean, const float* __restrict__ rstd,
                        __bf16* __restrict__ Ebf, __bf16* __restrict__ Cobf) {
  long idx = (long)blockIdx.x * blockDim.x + threadIdx.x;
  if (idx >= (long)BB * SS) return;
  int b  = (int)(idx >> 20);
  int ij = (int)(idx & (SS - 1));
  int i = ij >> 10, j = ij & (NPTS - 1);
  float g  = G[idx];
  float di = diag[b * NPTS + i], dj = diag[b * NPTS + j];
  float e  = sqrtf(fmaxf(di + dj - 2.f * g, 0.f));
  float cov  = g - (float)CD * mean[b * NPTS + i] * mean[b * NPTS + j];
  float corr = cov * rstd[b * NPTS + i] * rstd[b * NPTS + j];
  corr = fminf(1.f, fmaxf(-1.f, corr));
  Ebf[idx]  = (__bf16)e;
  Cobf[idx] = (__bf16)corr;
}

// ======================================================================
// K4: Chebyshev distance, 32x32 (i,j) tile per block via LDS
// ======================================================================
__global__ __launch_bounds__(256)
void k_cheby(const float* __restrict__ x, __bf16* __restrict__ Chbf) {
  __shared__ float xi[32][CD + 1];
  __shared__ float xj[32][CD + 1];
  int b  = blockIdx.z;
  int i0 = blockIdx.y * 32, j0 = blockIdx.x * 32;
  int t  = threadIdx.x;
  #pragma unroll
  for (int q = 0; q < 8; ++q) {
    int e = t + q * 256;
    int r = e >> 6, c = e & 63;
    xi[r][c] = x[((long)b * NPTS + i0 + r) * CD + c];
    xj[r][c] = x[((long)b * NPTS + j0 + r) * CD + c];
  }
  __syncthreads();
  #pragma unroll
  for (int q = 0; q < 4; ++q) {
    int p = t + q * 256;
    int li = p >> 5, lj = p & 31;
    float m = 0.f;
    #pragma unroll 8
    for (int c = 0; c < CD; ++c) m = fmaxf(m, fabsf(xi[li][c] - xj[lj][c]));
    Chbf[(long)b * SS + (long)(i0 + li) * NPTS + (j0 + lj)] = (__bf16)m;
  }
}

// ======================================================================
// K6: row softmax of logits*0.125 -> bf16 attn
// ======================================================================
__global__ __launch_bounds__(256)
void k_softmax(const float* __restrict__ L, __bf16* __restrict__ Attn) {
  __shared__ float red[256];
  long row = blockIdx.x;                       // 0 .. B*N-1
  const float* lr = L + row * (long)NPTS;
  int t = threadIdx.x;
  float vals[4], vmax = -3.0e38f;
  #pragma unroll
  for (int q = 0; q < 4; ++q) { vals[q] = lr[t + q * 256]; vmax = fmaxf(vmax, vals[q]); }
  red[t] = vmax; __syncthreads();
  for (int s = 128; s > 0; s >>= 1) { if (t < s) red[t] = fmaxf(red[t], red[t + s]); __syncthreads(); }
  vmax = red[0]; __syncthreads();
  float e4[4], se = 0.f;
  #pragma unroll
  for (int q = 0; q < 4; ++q) { e4[q] = __expf(0.125f * (vals[q] - vmax)); se += e4[q]; }
  red[t] = se; __syncthreads();
  for (int s = 128; s > 0; s >>= 1) { if (t < s) red[t] += red[t + s]; __syncthreads(); }
  float inv = 1.f / red[0];
  #pragma unroll
  for (int q = 0; q < 4; ++q) Attn[row * (long)NPTS + t + q * 256] = (__bf16)(e4[q] * inv);
}

// ======================================================================
// Top-K selection: exact threshold via radix histograms, stable ties
// ======================================================================
__device__ __forceinline__ unsigned f2u(float f) {
  unsigned u = __float_as_uint(f);
  return (u & 0x80000000u) ? ~u : (u | 0x80000000u);   // order-preserving
}
// sel layout per batch (8 ints): 0=binHi 1=cntGreaterHi 2=ut 3=kEq 4=cEq
//                                5=idxHiBin 6=remEq 7=idxThresh

__global__ void k_zero(unsigned* p, int n) {
  int i = blockIdx.x * blockDim.x + threadIdx.x;
  if (i < n) p[i] = 0u;
}

__global__ void k_hist_hi(const float* __restrict__ W, unsigned* __restrict__ h) {
  long idx = (long)blockIdx.x * blockDim.x + threadIdx.x;
  if (idx >= (long)BB * SS) return;
  int b = (int)(idx >> 20);
  unsigned u = f2u(W[idx]);
  atomicAdd(&h[(b << 16) + (u >> 16)], 1u);
}

__global__ void k_pick_hi(const unsigned* __restrict__ h, int* __restrict__ sel) {
  int b = threadIdx.x;
  if (b >= BB) return;
  const unsigned* hb = h + (b << 16);
  unsigned cum = 0;
  for (int bin = 65535; bin >= 0; --bin) {
    unsigned c = hb[bin];
    if (cum + c >= (unsigned)KTOP) { sel[b * 8 + 0] = bin; sel[b * 8 + 1] = (int)cum; return; }
    cum += c;
  }
}

__global__ void k_hist_lo(const float* __restrict__ W, const int* __restrict__ sel,
                          unsigned* __restrict__ h) {
  long idx = (long)blockIdx.x * blockDim.x + threadIdx.x;
  if (idx >= (long)BB * SS) return;
  int b = (int)(idx >> 20);
  unsigned u = f2u(W[idx]);
  if ((int)(u >> 16) == sel[b * 8 + 0]) atomicAdd(&h[(b << 16) + (u & 0xFFFFu)], 1u);
}

__global__ void k_pick_lo(const unsigned* __restrict__ h, int* __restrict__ sel) {
  int b = threadIdx.x;
  if (b >= BB) return;
  const unsigned* hb = h + (b << 16);
  unsigned cum = (unsigned)sel[b * 8 + 1];
  unsigned binHi = (unsigned)sel[b * 8 + 0];
  for (int bin = 65535; bin >= 0; --bin) {
    unsigned c = hb[bin];
    if (cum + c >= (unsigned)KTOP) {
      sel[b * 8 + 2] = (int)((binHi << 16) | (unsigned)bin);
      sel[b * 8 + 3] = (int)((unsigned)KTOP - cum);   // kEq >= 1
      sel[b * 8 + 4] = (int)c;                        // cEq
      return;
    }
    cum += c;
  }
}

__global__ void k_hist_ih(const float* __restrict__ W, const int* __restrict__ sel,
                          unsigned* __restrict__ h) {
  long idx = (long)blockIdx.x * blockDim.x + threadIdx.x;
  if (idx >= (long)BB * SS) return;
  int b = (int)(idx >> 20);
  if (f2u(W[idx]) == (unsigned)sel[b * 8 + 2]) {
    int ij = (int)(idx & (SS - 1));
    atomicAdd(&h[b * 1024 + (ij >> 10)], 1u);
  }
}

__global__ void k_pick_ih(const unsigned* __restrict__ h, int* __restrict__ sel) {
  int b = threadIdx.x;
  if (b >= BB) return;
  int kEq = sel[b * 8 + 3], cEq = sel[b * 8 + 4];
  if (kEq >= cEq) { sel[b * 8 + 5] = -1; sel[b * 8 + 7] = 0; return; }  // keep all ties
  const unsigned* hb = h + b * 1024;
  unsigned cum = 0;
  for (int bin = 1023; bin >= 0; --bin) {
    unsigned c = hb[bin];
    if (cum + c >= (unsigned)kEq) { sel[b * 8 + 5] = bin; sel[b * 8 + 6] = (int)((unsigned)kEq - cum); return; }
    cum += c;
  }
}

__global__ void k_hist_il(const float* __restrict__ W, const int* __restrict__ sel,
                          unsigned* __restrict__ h) {
  long idx = (long)blockIdx.x * blockDim.x + threadIdx.x;
  if (idx >= (long)BB * SS) return;
  int b = (int)(idx >> 20);
  if (f2u(W[idx]) == (unsigned)sel[b * 8 + 2]) {
    int ij = (int)(idx & (SS - 1));
    if ((ij >> 10) == sel[b * 8 + 5]) atomicAdd(&h[b * 1024 + (ij & 1023)], 1u);
  }
}

__global__ void k_pick_il(const unsigned* __restrict__ h, int* __restrict__ sel) {
  int b = threadIdx.x;
  if (b >= BB) return;
  int ihb = sel[b * 8 + 5];
  if (ihb < 0) return;                        // idxThresh already set
  const unsigned* hb = h + b * 1024;
  unsigned rem = (unsigned)sel[b * 8 + 6], cum = 0;
  for (int bin = 1023; bin >= 0; --bin) {
    unsigned c = hb[bin];
    if (cum + c >= rem) { sel[b * 8 + 7] = (ihb << 10) | bin; return; }
    cum += c;
  }
}

__global__ void k_mask(const float* __restrict__ W, const int* __restrict__ sel,
                       float* __restrict__ out) {
  long idx = (long)blockIdx.x * blockDim.x + threadIdx.x;
  if (idx >= (long)BB * SS) return;
  int b  = (int)(idx >> 20);
  int ij = (int)(idx & (SS - 1));
  unsigned u  = f2u(W[idx]);
  unsigned ut = (unsigned)sel[b * 8 + 2];
  int thr = sel[b * 8 + 7];
  out[idx] = (u > ut || (u == ut && ij >= thr)) ? 1.0f : 0.0f;
}

// ======================================================================
// launch
// ======================================================================
extern "C" void kernel_launch(void* const* d_in, const int* in_sizes, int n_in,
                              void* d_out, int out_size, void* d_ws, size_t ws_size,
                              hipStream_t stream) {
  (void)in_sizes; (void)n_in; (void)out_size; (void)ws_size;
  const float* x = (const float*)d_in[0];
  float* out = (float*)d_out;

  char* ws = (char*)d_ws;
  size_t o = 0;
  auto alloc = [&](size_t bytes) -> char* {
    char* p = ws + o;
    o = (o + bytes + 255) & ~(size_t)255;
    return p;
  };
  __bf16* xbf  = (__bf16*)alloc((size_t)BB * NPTS * CD * 2);
  float*  mean = (float*) alloc((size_t)BB * NPTS * 4);
  float*  diag = (float*) alloc((size_t)BB * NPTS * 4);
  float*  rstd = (float*) alloc((size_t)BB * NPTS * 4);
  float*  G    = (float*) alloc((size_t)BB * SS * 4);
  __bf16* Ebf  = (__bf16*)alloc((size_t)BB * SS * 2);   // E, later reused as attn
  __bf16* Chbf = (__bf16*)alloc((size_t)BB * SS * 2);
  __bf16* Cobf = (__bf16*)alloc((size_t)BB * SS * 2);
  float*  Lb   = (float*) alloc((size_t)BB * SS * 4);   // logits, later W
  unsigned* histHi = (unsigned*)alloc((size_t)BB * 65536 * 4);  // contiguous with:
  unsigned* histLo = (unsigned*)alloc((size_t)BB * 65536 * 4);
  unsigned* histIH = (unsigned*)alloc((size_t)BB * 1024 * 4);
  unsigned* histIL = (unsigned*)alloc((size_t)BB * 1024 * 4);
  int*      sel    = (int*)     alloc((size_t)BB * 8 * 4);

  // 1) row stats + bf16 x
  k_rowstats<<<(BB * NPTS + 255) / 256, 256, 0, stream>>>(x, xbf, mean);

  dim3 gg(NPTS / 128, NPTS / 128, BB);
  // 2) Gram: G = x . x^T  (K=64)
  k_gemm_nt<<<gg, 256, 0, stream>>>(xbf, xbf, G, NPTS, NPTS, CD,
                                    (long)NPTS * CD, (long)NPTS * CD, (long)SS);
  // 3) per-row diag/std, then E and Corr
  k_rowaux<<<(BB * NPTS + 255) / 256, 256, 0, stream>>>(G, mean, diag, rstd);
  k_ecorr<<<(int)(((long)BB * SS + 255) / 256), 256, 0, stream>>>(G, diag, mean, rstd, Ebf, Cobf);
  // 4) Chebyshev
  dim3 gc(NPTS / 32, NPTS / 32, BB);
  k_cheby<<<gc, 256, 0, stream>>>(x, Chbf);
  // 5) logits = E . Ch^T (K=1024)
  k_gemm_nt<<<gg, 256, 0, stream>>>(Ebf, Chbf, Lb, NPTS, NPTS, NPTS,
                                    (long)SS, (long)SS, (long)SS);
  // 6) softmax -> attn (bf16, reuses E buffer)
  k_softmax<<<BB * NPTS, 256, 0, stream>>>(Lb, Ebf);
  // 7) W = attn . Corr^T (Corr symmetric => == attn @ Corr), into Lb
  k_gemm_nt<<<gg, 256, 0, stream>>>(Ebf, Cobf, Lb, NPTS, NPTS, NPTS,
                                    (long)SS, (long)SS, (long)SS);
  // 8) exact top-K binarization
  int histTotal = BB * 65536 * 2 + BB * 1024 * 2;   // histHi..histIL contiguous
  k_zero<<<(histTotal + 255) / 256, 256, 0, stream>>>(histHi, histTotal);
  int gb = (int)(((long)BB * SS + 255) / 256);
  k_hist_hi<<<gb, 256, 0, stream>>>(Lb, histHi);
  k_pick_hi<<<1, 32, 0, stream>>>(histHi, sel);
  k_hist_lo<<<gb, 256, 0, stream>>>(Lb, sel, histLo);
  k_pick_lo<<<1, 32, 0, stream>>>(histLo, sel);
  k_hist_ih<<<gb, 256, 0, stream>>>(Lb, sel, histIH);
  k_pick_ih<<<1, 32, 0, stream>>>(histIH, sel);
  k_hist_il<<<gb, 256, 0, stream>>>(Lb, sel, histIL);
  k_pick_il<<<1, 32, 0, stream>>>(histIL, sel);
  k_mask<<<gb, 256, 0, stream>>>(Lb, sel, out);
}
